// GraphNativeBrainModel_18949395710196
// MI455X (gfx1250) — matrix operand order repaired and checked by reading, not compiled, not run on hardware
//
#include <hip/hip_runtime.h>
#include <stdint.h>
#include <math.h>

#define N_NODES   20000
#define T_STEPS   10
#define H_DIM     64
#define E_EDGES   200000
#define L_LAYERS  2
#define ETY_N     2
#define LRELU_SLOPE 0.2f
#define BN_SCALE_C  0.9999950000374997f  // 1/sqrt(1+1e-5)

typedef __attribute__((ext_vector_type(16))) __bf16 v16bf;
typedef __attribute__((ext_vector_type(8)))  float  v8f;

union BFrag { v16bf v; uint32_t u[8]; };
union CFrag { v8f   v; float    f[8]; };

__device__ __forceinline__ unsigned short f2bf(float x) {
    uint32_t u = __float_as_uint(x);
    uint32_t r = (u + 0x7FFFu + ((u >> 16) & 1u)) >> 16;   // RNE
    return (unsigned short)r;
}
__device__ __forceinline__ float bf2f(unsigned short h) {
    return __uint_as_float(((uint32_t)h) << 16);
}

// CDNA5 async global->LDS bulk copy (ASYNCcnt-tracked), per-lane 128-bit.
__device__ __forceinline__ void async_ld_b128(uint32_t lds_addr, const void* gaddr) {
    asm volatile("global_load_async_to_lds_b128 %0, %1, off"
                 :: "v"(lds_addr), "v"(gaddr) : "memory");
}
__device__ __forceinline__ void wait_async0() {
    asm volatile("s_wait_asynccnt 0x0" ::: "memory");
}

// Map (k,col) of a [K x COUT] B matrix to the swizzled-LDS fragment layout:
// fragment f = (k/32)*COLT + col/16; within f, lane = 16*((k>>3)&1) + (col&15)
// owns 16 contiguous bf16 (= 32B); element = (k&7) + 8*((k>>4)&1).
__device__ __forceinline__ int swz_idx(int k, int col, int COLT) {
    int ks = k >> 5, half = (k >> 3) & 1, rem = (k & 7) | (((k >> 4) & 1) << 3);
    int ct = col >> 4, l15 = col & 15;
    int f = ks * COLT + ct;
    int lane = half * 16 + l15;
    return ((f * 32 + lane) << 4) + rem;
}

// ---------------------------------------------------------------- utilities
__global__ void k_fill(float* __restrict__ p, float v, int n) {
    int i = blockIdx.x * blockDim.x + threadIdx.x;
    if (i < n) p[i] = v;
}

__global__ void k_init_x(const float* __restrict__ x, float* __restrict__ xc,
                         unsigned short* __restrict__ xb, int n) {
    int i = blockIdx.x * blockDim.x + threadIdx.x;
    if (i < n) { float v = x[i]; xc[i] = v; xb[i] = f2bf(v); }
}

// ------------------------------------------------ h = x@W + b, s = lrelu(h.a)
// One 16-row tile per wave; async-staged A tile, swizzled B frags (ds_load_b128).
__global__ __launch_bounds__(256)
void k_gemm_score(const unsigned short* __restrict__ xbf,
                  const float* __restrict__ W, const float* __restrict__ bias,
                  const float* __restrict__ avec,
                  float* __restrict__ Hout, float* __restrict__ scout, int rows) {
    __shared__ unsigned short wswz[2 * 4 * 32 * 16];   // 8KB swizzled weights
    __shared__ unsigned short atile[8][16 * 64];       // 16KB: per-wave A tiles

    // stage W (coalesced reads, swizzled bf16 writes)
    for (int sidx = threadIdx.x; sidx < H_DIM * H_DIM; sidx += blockDim.x) {
        int k = sidx >> 6, col = sidx & 63;            // W[k][col], row-major
        wswz[swz_idx(k, col, 4)] = f2bf(W[sidx]);
    }
    __syncthreads();

    const int lane = threadIdx.x & 31;
    const int wave = threadIdx.x >> 5;
    const int half = lane >> 4;
    const int l15  = lane & 15;

    int m0 = (blockIdx.x * 8 + wave) * 16;
    if (m0 >= rows) return;

    // async bulk copy of the 16x64 bf16 A tile into this wave's LDS window
    const unsigned short* tbase = xbf + (size_t)m0 * H_DIM;
    uint32_t awb = (uint32_t)(uintptr_t)(&atile[wave][0]);
    for (int c = lane; c < 128; c += 32) {             // 128 x 16B chunks
        int rw = c >> 3, part = c & 7;
        async_ld_b128(awb + rw * 128 + part * 16, tbase + rw * 64 + part * 8);
    }
    wait_async0();

    float bias_r[4], a_r[4];
#pragma unroll
    for (int ct = 0; ct < 4; ct++) {
        int col = ct * 16 + l15;
        bias_r[ct] = bias[col];
        a_r[ct]    = avec[col];
    }

    CFrag acc[4];
#pragma unroll
    for (int ct = 0; ct < 4; ct++)
#pragma unroll
        for (int j = 0; j < 8; j++) acc[ct].f[j] = 0.f;

    const unsigned short* aw = &atile[wave][0];
#pragma unroll
    for (int ks = 0; ks < 2; ks++) {
        BFrag af;  // A 16x32 bf16: pair v -> K = (v<4 ? 2v : 16+2(v-4)) + 8*half
#pragma unroll
        for (int v = 0; v < 8; v++) {
            int kk = ((v < 4) ? 2 * v : 16 + 2 * (v - 4)) + 8 * half + 32 * ks;
            af.u[v] = *reinterpret_cast<const uint32_t*>(aw + l15 * 64 + kk);
        }
#pragma unroll
        for (int ct = 0; ct < 4; ct++) {
            const uint32_t* bp = reinterpret_cast<const uint32_t*>(
                wswz + (((ks * 4 + ct) * 32 + lane) << 4));
            BFrag bfr;
#pragma unroll
            for (int v = 0; v < 8; v++) bfr.u[v] = bp[v];   // 2x ds_load_b128
            acc[ct].v = __builtin_amdgcn_wmma_f32_16x16x32_bf16(
                false, af.v, false, bfr.v, (short)0, acc[ct].v, false, false);
        }
    }

    // bias add, store H, per-row dot with attention vector a
    float p[8];
#pragma unroll
    for (int vi = 0; vi < 8; vi++) p[vi] = 0.f;
#pragma unroll
    for (int ct = 0; ct < 4; ct++) {
        int col = ct * 16 + l15;
#pragma unroll
        for (int vi = 0; vi < 8; vi++) {
            float hv = acc[ct].f[vi] + bias_r[ct];
            Hout[(size_t)(m0 + vi + 8 * half) * H_DIM + col] = hv;
            p[vi] += hv * a_r[ct];
        }
    }
#pragma unroll
    for (int off = 1; off < 16; off <<= 1)
#pragma unroll
        for (int vi = 0; vi < 8; vi++) p[vi] += __shfl_xor(p[vi], off, 32);
    if (l15 == 0) {
#pragma unroll
        for (int vi = 0; vi < 8; vi++) {
            float s = p[vi];
            s = (s > 0.f) ? s : LRELU_SLOPE * s;
            scout[m0 + vi + 8 * half] = s;
        }
    }
}

// ------------------------------------------------------- edge pass 1: seg max
__device__ __forceinline__ void atomicMaxFloat(float* addr, float v) {
    if (v >= 0.f) atomicMax((int*)addr, __float_as_int(v));
    else          atomicMin((unsigned int*)addr, __float_as_uint(v));
}

__global__ void k_edge_pass1(const int* __restrict__ src, const int* __restrict__ dst,
                             const float* __restrict__ sc, float* __restrict__ smax, int ET) {
    int i = blockIdx.x * blockDim.x + threadIdx.x;
    if (i >= ET) return;
    int e = i / T_STEPS, t = i - e * T_STEPS;
    float s = sc[src[e] * T_STEPS + t];
    atomicMaxFloat(smax + dst[e] * T_STEPS + t, s);
}

// ------------------ edge pass 2: ex, den scatter, unnormalized weighted gather
__global__ void k_edge_pass2(const int* __restrict__ src, const int* __restrict__ dst,
                             const float* __restrict__ ea, const float* __restrict__ sc,
                             const float* __restrict__ smax, const float* __restrict__ Hbuf,
                             float* __restrict__ den, float* __restrict__ accb, int total) {
    const int stride = gridDim.x * blockDim.x;
    for (int i = blockIdx.x * blockDim.x + threadIdx.x; i < total; i += stride) {
        int nx = i + stride;                            // prefetch next gather
        if (nx < total) {
            int et2 = nx >> 6;
            int e2 = et2 / T_STEPS, t2 = et2 - e2 * T_STEPS;
            __builtin_prefetch(Hbuf + (size_t)(src[e2] * T_STEPS + t2) * 64 + (nx & 63), 0, 1);
        }
        int h  = i & 63;
        int et = i >> 6;
        int e  = et / T_STEPS, t = et - e * T_STEPS;
        int rs = src[e] * T_STEPS + t;
        int rd = dst[e] * T_STEPS + t;
        float ex = __expf(sc[rs] - smax[rd]);
        if (h == 0) atomicAdd(den + rd, ex);
        float w = ex * ea[e];
        atomicAdd(accb + (size_t)rd * 64 + h, w * Hbuf[(size_t)rs * 64 + h]);
    }
}

// ------------------------------------------- msg += acc/(den+eps) (per type)
__global__ void k_norm_acc(const float* __restrict__ accb, const float* __restrict__ den,
                           float* __restrict__ msg, int total) {
    int i = blockIdx.x * blockDim.x + threadIdx.x;
    if (i < total) {
        int row = i >> 6;
        msg[i] += accb[i] / (den[row] + 1e-16f);
    }
}

// --------------------------- x = LN(x + msg/ETY); wave32 row reduction, H=64
__global__ __launch_bounds__(256)
void k_node_update(float* __restrict__ xc, unsigned short* __restrict__ xb,
                   const float* __restrict__ msg, const float* __restrict__ g,
                   const float* __restrict__ b, int rowsNT) {
    int lane = threadIdx.x & 31, wave = threadIdx.x >> 5;
    int row = blockIdx.x * 8 + wave;
    if (row >= rowsNT) return;
    size_t base = (size_t)row * 64;
    float v0 = xc[base + lane]      + 0.5f * msg[base + lane];
    float v1 = xc[base + 32 + lane] + 0.5f * msg[base + 32 + lane];
    float s = v0 + v1, sq = v0 * v0 + v1 * v1;
#pragma unroll
    for (int off = 1; off < 32; off <<= 1) {
        s  += __shfl_xor(s,  off, 32);
        sq += __shfl_xor(sq, off, 32);
    }
    float mean = s * (1.f / 64.f);
    float var  = sq * (1.f / 64.f) - mean * mean;
    float inv  = rsqrtf(var + 1e-5f);
    float o0 = (v0 - mean) * inv * g[lane]      + b[lane];
    float o1 = (v1 - mean) * inv * g[32 + lane] + b[32 + lane];
    xc[base + lane] = o0;       xc[base + 32 + lane] = o1;
    xb[base + lane] = f2bf(o0); xb[base + 32 + lane] = f2bf(o1);
}

// ---------- 3-tap temporal conv as implicit GEMM (K=192) + BN(eval) + ReLU
// Async-staged 18-row halo window per wave handles t-boundary zero padding.
template<int COLT>
__global__ __launch_bounds__(256)
void k_conv_gemm(const unsigned short* __restrict__ inbf, const float* __restrict__ w,
                 const float* __restrict__ cb, const float* __restrict__ bng,
                 const float* __restrict__ bnb, unsigned short* __restrict__ outbf, int rows) {
    const int COUT = COLT * 16;
    __shared__ unsigned short wswz[6 * COLT * 32 * 16];  // swizzled weights
    __shared__ unsigned short awin[8][18 * 64];          // 18-row halo windows

    // stage weights: w[o][i][dt] -> B[k=dt*64+i][o], swizzled bf16
    for (int sidx = threadIdx.x; sidx < 192 * COUT; sidx += blockDim.x) {
        int o = sidx / 192, r = sidx - o * 192;
        int ii = r / 3, dt = r - ii * 3;
        wswz[swz_idx(dt * 64 + ii, o, COLT)] = f2bf(w[sidx]);
    }
    __syncthreads();

    const int lane = threadIdx.x & 31, wave = threadIdx.x >> 5;
    const int half = lane >> 4, l15 = lane & 15;
    int m0 = (blockIdx.x * 8 + wave) * 16;
    if (m0 >= rows) return;

    // async copy rows [m0-1, m0+16] (clamped) into this wave's halo window
    uint32_t awb = (uint32_t)(uintptr_t)(&awin[wave][0]);
    for (int c = lane; c < 144; c += 32) {               // 18 rows x 8 chunks
        int rw = c >> 3, part = c & 7;
        int gr = m0 - 1 + rw;
        if (gr >= 0 && gr < rows)
            async_ld_b128(awb + rw * 128 + part * 16,
                          inbf + (size_t)gr * 64 + part * 8);
    }
    wait_async0();

    int row = m0 + l15;
    int n = row / T_STEPS, t = row - n * T_STEPS;

    CFrag acc[COLT];
#pragma unroll
    for (int ct = 0; ct < COLT; ct++)
#pragma unroll
        for (int j = 0; j < 8; j++) acc[ct].f[j] = 0.f;

    float cb_r[COLT], bg_r[COLT], bb_r[COLT];
#pragma unroll
    for (int ct = 0; ct < COLT; ct++) {
        int col = ct * 16 + l15;
        cb_r[ct] = cb[col];
        bg_r[ct] = bng[col] * BN_SCALE_C;
        bb_r[ct] = bnb[col];
    }

    const unsigned short* aw = &awin[wave][0];
#pragma unroll
    for (int ks = 0; ks < 6; ks++) {
        BFrag af;
#pragma unroll
        for (int v = 0; v < 8; v++) {
            int kk = ((v < 4) ? 2 * v : 16 + 2 * (v - 4)) + 8 * half + 32 * ks;
            int dt = kk >> 6, ii = kk & 63;
            int tt = t + dt - 1;
            uint32_t u = 0;
            if (tt >= 0 && tt < T_STEPS)   // window row = l15 + dt
                u = *reinterpret_cast<const uint32_t*>(aw + (l15 + dt) * 64 + ii);
            af.u[v] = u;
        }
#pragma unroll
        for (int ct = 0; ct < COLT; ct++) {
            const uint32_t* bp = reinterpret_cast<const uint32_t*>(
                wswz + (((ks * COLT + ct) * 32 + lane) << 4));
            BFrag bfr;
#pragma unroll
            for (int v = 0; v < 8; v++) bfr.u[v] = bp[v];
            acc[ct].v = __builtin_amdgcn_wmma_f32_16x16x32_bf16(
                false, af.v, false, bfr.v, (short)0, acc[ct].v, false, false);
        }
    }

#pragma unroll
    for (int ct = 0; ct < COLT; ct++) {
        int col = ct * 16 + l15;
#pragma unroll
        for (int vi = 0; vi < 8; vi++) {
            float y = (acc[ct].f[vi] + cb_r[ct]) * bg_r[ct] + bb_r[ct];
            y = y > 0.f ? y : 0.f;
            outbf[(size_t)(m0 + vi + 8 * half) * COUT + col] = f2bf(y);
        }
    }
}

// ------------------------------------------------- conv3: 32 ch -> 1, k=3
__global__ void k_conv3(const unsigned short* __restrict__ y2, const float* __restrict__ w3,
                        const float* __restrict__ cb3, float* __restrict__ out, int rows) {
    int row = blockIdx.x * blockDim.x + threadIdx.x;
    if (row >= rows) return;
    int n = row / T_STEPS, t = row - n * T_STEPS;
    float acc = cb3[0];
#pragma unroll
    for (int dt = 0; dt < 3; dt++) {
        int tt = t + dt - 1;
        if (tt < 0 || tt >= T_STEPS) continue;
        const uint32_t* p = reinterpret_cast<const uint32_t*>(y2 + (size_t)(n * T_STEPS + tt) * 32);
#pragma unroll
        for (int j = 0; j < 16; j++) {
            uint32_t u = p[j];
            acc += bf2f((unsigned short)(u & 0xFFFF)) * w3[(2 * j) * 3 + dt];
            acc += bf2f((unsigned short)(u >> 16))    * w3[(2 * j + 1) * 3 + dt];
        }
    }
    out[row] = acc;
}

// ---------------------------------------------------------------- launcher
extern "C" void kernel_launch(void* const* d_in, const int* in_sizes, int n_in,
                              void* d_out, int out_size, void* d_ws, size_t ws_size,
                              hipStream_t stream) {
    const float* x     = (const float*)d_in[0];
    const int*   ei    = (const int*)  d_in[1];
    const float* eattr = (const float*)d_in[2];
    const float* Wp    = (const float*)d_in[3];
    const float* bp    = (const float*)d_in[4];
    const float* ap    = (const float*)d_in[5];
    const float* lng   = (const float*)d_in[6];
    const float* lnb   = (const float*)d_in[7];
    const float* cw1   = (const float*)d_in[8];
    const float* cb1   = (const float*)d_in[9];
    const float* bn1g  = (const float*)d_in[10];
    const float* bn1b  = (const float*)d_in[11];
    const float* cw2   = (const float*)d_in[12];
    const float* cb2   = (const float*)d_in[13];
    const float* bn2g  = (const float*)d_in[14];
    const float* bn2b  = (const float*)d_in[15];
    const float* cw3   = (const float*)d_in[16];
    const float* cb3   = (const float*)d_in[17];

    const int rowsNT = N_NODES * T_STEPS;   // 200000
    const int elems  = rowsNT * H_DIM;      // 12.8M

    char* ws = (char*)d_ws;
    size_t off = 0;
    float*          x_cur = (float*)(ws + off);          off += (size_t)elems * 4;
    unsigned short* xbf   = (unsigned short*)(ws + off); off += (size_t)elems * 2;
    float*          Hbuf  = (float*)(ws + off);          off += (size_t)elems * 4;
    float*          accb  = (float*)(ws + off);          off += (size_t)elems * 4;
    float*          msg   = (float*)(ws + off);          off += (size_t)elems * 4;
    float*          sc    = (float*)(ws + off);          off += (size_t)rowsNT * 4;
    float*          smax  = (float*)(ws + off);          off += (size_t)rowsNT * 4;
    float*          den   = (float*)(ws + off);          off += (size_t)rowsNT * 4;
    unsigned short* y1bf  = (unsigned short*)Hbuf;   // decoder reuses retired buffers
    unsigned short* y2bf  = (unsigned short*)accb;

    dim3 blk(256);
    auto g1 = [](int n) { return dim3((unsigned)((n + 255) / 256)); };

    k_init_x<<<g1(elems), blk, 0, stream>>>(x, x_cur, xbf, elems);

    const int gemmBlocks = (rowsNT / 16 + 7) / 8;  // 12500 tiles, 8 waves/block

    for (int l = 0; l < L_LAYERS; l++) {
        k_fill<<<g1(elems), blk, 0, stream>>>(msg, 0.f, elems);
        for (int e = 0; e < ETY_N; e++) {
            const float* W  = Wp + ((size_t)(l * ETY_N + e)) * H_DIM * H_DIM;
            const float* bb = bp + ((size_t)(l * ETY_N + e)) * H_DIM;
            const float* aa = ap + ((size_t)(l * ETY_N + e)) * H_DIM;
            k_gemm_score<<<gemmBlocks, blk, 0, stream>>>(xbf, W, bb, aa, Hbuf, sc, rowsNT);

            k_fill<<<g1(rowsNT), blk, 0, stream>>>(smax, -INFINITY, rowsNT);
            k_fill<<<g1(rowsNT), blk, 0, stream>>>(den, 0.f, rowsNT);
            k_fill<<<g1(elems),  blk, 0, stream>>>(accb, 0.f, elems);

            const int* srcA = ei + (size_t)e * 2 * E_EDGES;
            const int* dstA = srcA + E_EDGES;
            const float* eaA = eattr + (size_t)e * E_EDGES;
            const int ET = E_EDGES * T_STEPS;          // 2M
            k_edge_pass1<<<g1(ET), blk, 0, stream>>>(srcA, dstA, sc, smax, ET);
            const int tot = ET * 64;                   // 128M
            k_edge_pass2<<<g1(tot / 2), blk, 0, stream>>>(srcA, dstA, eaA, sc, smax,
                                                          Hbuf, den, accb, tot);
            k_norm_acc<<<g1(elems), blk, 0, stream>>>(accb, den, msg, elems);
        }
        k_node_update<<<(rowsNT + 7) / 8, blk, 0, stream>>>(
            x_cur, xbf, msg, lng + (size_t)l * H_DIM, lnb + (size_t)l * H_DIM, rowsNT);
    }

    k_conv_gemm<4><<<gemmBlocks, blk, 0, stream>>>(xbf,  cw1, cb1, bn1g, bn1b, y1bf, rowsNT);
    k_conv_gemm<2><<<gemmBlocks, blk, 0, stream>>>(y1bf, cw2, cb2, bn2g, bn2b, y2bf, rowsNT);
    k_conv3<<<g1(rowsNT), blk, 0, stream>>>(y2bf, cw3, cb3, (float*)d_out, rowsNT);
}
